// LlamaAttention_heavy_hitter_74182675136969
// MI455X (gfx1250) — compile-verified, matrix operand order checked by reading, not run on hardware
//
#include <hip/hip_runtime.h>

#define S_LEN 2048
#define HID   2048
#define NH    16
#define NKV   4
#define DH    128
#define HEAVY 204
#define RECENT 204

typedef __attribute__((ext_vector_type(16))) _Float16     v16h;
typedef __attribute__((ext_vector_type(8)))  float        v8f;
typedef __attribute__((ext_vector_type(4)))  unsigned int v4u;
typedef __attribute__((ext_vector_type(8)))  unsigned int v8u;

// ---------------------------------------------------------------------------
// Fragment loader for 16-bit A (16xK row-major) / B-transposed (NxK) tiles.
// CDNA5 16-bit A layout (16x32):
//   lanes 0-15 : row M = lane,    halves [0..7]=K0..7,  [8..15]=K16..23
//   lanes 16-31: row M = lane-16, halves [0..7]=K8..15, [8..15]=K24..31
// `stride` in halves; rows must be 16B aligned.
// ---------------------------------------------------------------------------
__device__ __forceinline__ v16h frag_ld(const _Float16* base, int stride) {
  const int lane = threadIdx.x & 31;
  const int row  = lane & 15;
  const int ksel = (lane >> 4) & 1;
  const _Float16* p = base + row * stride + ksel * 8;
  union { v16h v; float4 q[2]; } f;
  f.q[0] = *(const float4*)(p);
  f.q[1] = *(const float4*)(p + 16);
  return f.v;
}

__device__ __forceinline__ v8f wmma32(v16h a, v16h b, v8f c) {
  return __builtin_amdgcn_wmma_f32_16x16x32_f16(false, a, false, b, (short)0, c,
                                                false, false);
}

// ---------------------------------------------------------------------------
// C(MxN,f32) = A(MxK,f16) * B(KxN,f16).  Block tile 128x128, 8 waves * 64x32.
// A tile staged via GLOBAL_LOAD_ASYNC_TO_LDS_B128 (ASYNCcnt path, padded LDS
// destinations supplied per lane); B tile transposed through VGPRs.
// ---------------------------------------------------------------------------
__global__ __launch_bounds__(256) void gemm_wmma(const _Float16* __restrict__ A,
                                                 const _Float16* __restrict__ B,
                                                 float* __restrict__ C,
                                                 int M, int N, int K) {
  __shared__ _Float16 As[128][40];   // stride 40 halves = 80B (16B multiple)
  __shared__ _Float16 Bs[128][40];   // B transposed: Bs[n][k]
  const int bm = blockIdx.y * 128, bn = blockIdx.x * 128;
  const int tid = threadIdx.x, wave = tid >> 5, lane = tid & 31;
  const int wm = (wave >> 2) * 64, wn = (wave & 3) * 32;
  const unsigned int as_lds = (unsigned int)(unsigned long long)&As[0][0];
  v8f acc[4][2] = {};
  for (int k0 = 0; k0 < K; k0 += 32) {
    // --- A tile 128x32 halves: async copy global->LDS, 2x16B per thread ---
#pragma unroll
    for (int i = 0; i < 2; ++i) {
      const int seg = tid * 2 + i;           // 512 segments of 16B
      const int row = seg >> 2, s4 = seg & 3;
      const unsigned int dst = as_lds + (unsigned int)(row * 80 + s4 * 16);
      const unsigned long long ga = (unsigned long long)
          (A + (size_t)(bm + row) * K + (k0 + s4 * 8));
      asm volatile("global_load_async_to_lds_b128 %0, %1, off"
                   :: "v"(dst), "v"(ga) : "memory");
    }
    // --- B tile 32x128 halves -> Bs[n][k] (transpose via VGPRs) ---
    {
      const int r = tid >> 3, cb = (tid & 7) * 16;
      const _Float16* srcB = B + (size_t)(k0 + r) * N + (bn + cb);
      union { float4 q; _Float16 h[8]; } b0, b1;
      b0.q = *(const float4*)(srcB);
      b1.q = *(const float4*)(srcB + 8);
#pragma unroll
      for (int c = 0; c < 8; ++c) {
        Bs[cb + c][r]     = b0.h[c];
        Bs[cb + 8 + c][r] = b1.h[c];
      }
    }
    asm volatile("s_wait_asynccnt 0x0" ::: "memory");
    __syncthreads();
    const v16h b0 = frag_ld(&Bs[wn][0], 40);
    const v16h b1 = frag_ld(&Bs[wn + 16][0], 40);
#pragma unroll
    for (int mt = 0; mt < 4; ++mt) {
      const v16h a = frag_ld(&As[wm + mt * 16][0], 40);
      acc[mt][0] = wmma32(a, b0, acc[mt][0]);
      acc[mt][1] = wmma32(a, b1, acc[mt][1]);
    }
    __syncthreads();
  }
  // C-frag: VGPR r -> (M = r + 8*(lane>=16), N = lane&15)
  const int nlo = lane & 15, hi8 = (lane >> 4) * 8;
#pragma unroll
  for (int mt = 0; mt < 4; ++mt)
#pragma unroll
    for (int nt = 0; nt < 2; ++nt) {
      union { v8f v; float f[8]; } u; u.v = acc[mt][nt];
      float* cp = C + (size_t)(bm + wm + mt * 16 + hi8) * N
                    + (bn + wn + nt * 16 + nlo);
#pragma unroll
      for (int r = 0; r < 8; ++r) { *cp = u.f[r]; cp += N; }
    }
}

// ---------------------------------------------------------------------------
// fp32 -> f16 conversion (for X and the four weight matrices)
// ---------------------------------------------------------------------------
__global__ __launch_bounds__(256) void f32_to_f16(const float* __restrict__ s,
                                                  _Float16* __restrict__ d, int n) {
  const int i = blockIdx.x * 256 + threadIdx.x;
  if (i < n) d[i] = (_Float16)s[i];
}

// ---------------------------------------------------------------------------
// RoPE + fp32->f16 + transpose to [H][S][D]
// ---------------------------------------------------------------------------
__global__ __launch_bounds__(256) void rope_q(const float* __restrict__ q,
                                              _Float16* __restrict__ q16) {
  const int idx = blockIdx.x * 256 + threadIdx.x;
  const int d = idx & 63, h = (idx >> 6) & (NH - 1), s = idx >> 10;
  const float* src = q + (size_t)s * HID + h * DH;
  const float x1 = src[d], x2 = src[d + 64];
  const float f = __powf(10000.0f, -(float)d * (1.0f / 64.0f));
  float sn, cs;
  __sincosf((float)s * f, &sn, &cs);
  _Float16* dst = q16 + ((size_t)h * S_LEN + s) * DH;
  dst[d]      = (_Float16)(x1 * cs - x2 * sn);
  dst[d + 64] = (_Float16)(x2 * cs + x1 * sn);
}

__global__ __launch_bounds__(256) void rope_k(const float* __restrict__ k,
                                              _Float16* __restrict__ k16) {
  const int idx = blockIdx.x * 256 + threadIdx.x;
  const int d = idx & 63, h = (idx >> 6) & (NKV - 1), s = idx >> 8;
  const float* src = k + (size_t)s * (NKV * DH) + h * DH;
  const float x1 = src[d], x2 = src[d + 64];
  const float f = __powf(10000.0f, -(float)d * (1.0f / 64.0f));
  float sn, cs;
  __sincosf((float)s * f, &sn, &cs);
  _Float16* dst = k16 + ((size_t)h * S_LEN + s) * DH;
  dst[d]      = (_Float16)(x1 * cs - x2 * sn);
  dst[d + 64] = (_Float16)(x2 * cs + x1 * sn);
}

// V converted to f16 and stored TRANSPOSED: v16t[kv][d][s]  (feeds the TDM)
__global__ __launch_bounds__(256) void conv_v(const float* __restrict__ v,
                                              _Float16* __restrict__ v16t) {
  const int idx = blockIdx.x * 256 + threadIdx.x;
  const int d = idx & (DH - 1), h = (idx >> 7) & (NKV - 1), s = idx >> 9;
  v16t[((size_t)h * DH + d) * S_LEN + s] =
      (_Float16)v[(size_t)s * (NKV * DH) + h * DH + d];
}

// ---------------------------------------------------------------------------
// Oracle pass: causal softmax row sums, then colsum += exp/Z (H2O statistic).
// ---------------------------------------------------------------------------
__global__ __launch_bounds__(256) void attn_oracle(const _Float16* __restrict__ q16,
                                                   const _Float16* __restrict__ k16,
                                                   float* __restrict__ colsum) {
  const int h = blockIdx.x;
  const int wave = threadIdx.x >> 5, lane = threadIdx.x & 31;
  const int qt = blockIdx.y * 8 + wave, q0 = qt * 16;
  const int kv = h >> 2;
  const int nlo = lane & 15, hi8 = (lane >> 4) * 8;
  const float scale = 0.088388347648318447f;  // 1/sqrt(128)
  const _Float16* qptr = q16 + ((size_t)h * S_LEN + q0) * DH;
  v16h qa[4];
#pragma unroll
  for (int c = 0; c < 4; ++c) qa[c] = frag_ld(qptr + c * 32, DH);

  float rz[8] = {0, 0, 0, 0, 0, 0, 0, 0};
  for (int kt = 0; kt <= qt; ++kt) {
    const _Float16* kptr = k16 + ((size_t)kv * S_LEN + kt * 16) * DH;
    v8f s = {};
#pragma unroll
    for (int c = 0; c < 4; ++c) s = wmma32(qa[c], frag_ld(kptr + c * 32, DH), s);
    const int j = kt * 16 + nlo;
    union { v8f v; float f[8]; } u; u.v = s;
#pragma unroll
    for (int r = 0; r < 8; ++r) {
      const int i = q0 + hi8 + r;
      rz[r] += (j <= i) ? __expf(u.f[r] * scale) : 0.0f;
    }
  }
#pragma unroll
  for (int r = 0; r < 8; ++r) {
    rz[r] += __shfl_xor(rz[r], 1, 32);
    rz[r] += __shfl_xor(rz[r], 2, 32);
    rz[r] += __shfl_xor(rz[r], 4, 32);
    rz[r] += __shfl_xor(rz[r], 8, 32);
  }
  for (int kt = 0; kt <= qt; ++kt) {
    const _Float16* kptr = k16 + ((size_t)kv * S_LEN + kt * 16) * DH;
    v8f s = {};
#pragma unroll
    for (int c = 0; c < 4; ++c) s = wmma32(qa[c], frag_ld(kptr + c * 32, DH), s);
    const int j = kt * 16 + nlo;
    union { v8f v; float f[8]; } u; u.v = s;
    float cs = 0.0f;
#pragma unroll
    for (int r = 0; r < 8; ++r) {
      const int i = q0 + hi8 + r;
      cs += (j <= i) ? __expf(u.f[r] * scale) / rz[r] : 0.0f;
    }
    atomicAdd(&colsum[h * S_LEN + j], cs);
  }
}

// ---------------------------------------------------------------------------
// Per-head bitonic sort of colsum -> threshold = HEAVY-th largest value.
// ---------------------------------------------------------------------------
__global__ __launch_bounds__(256) void topk_thresh(const float* __restrict__ colsum,
                                                   float* __restrict__ thresh) {
  __shared__ float v[S_LEN];
  const int h = blockIdx.x;
  for (int i = threadIdx.x; i < S_LEN; i += 256) v[i] = colsum[h * S_LEN + i];
  __syncthreads();
  for (int k = 2; k <= S_LEN; k <<= 1)
    for (int j = k >> 1; j > 0; j >>= 1) {
      for (int t = threadIdx.x; t < S_LEN; t += 256) {
        const int ixj = t ^ j;
        if (ixj > t) {
          const bool up = ((t & k) == 0);
          const float a = v[t], b = v[ixj];
          if ((a > b) == up) { v[t] = b; v[ixj] = a; }
        }
      }
      __syncthreads();
    }
  if (threadIdx.x == 0) thresh[h] = v[S_LEN - HEAVY];
}

// ---------------------------------------------------------------------------
// Final pass: V^T tile (32 keys x 128 dims) staged by the Tensor Data Mover.
// The D# uses pad_interval=16 DWORDs + pad_amount=4 DWORDs so the TDM writes
// the 80B-padded LDS rows the fragment loader expects.  Wave 0 issues,
// waits TENSORcnt, block barriers.
// ---------------------------------------------------------------------------
__global__ __launch_bounds__(256) void attn_pv(const _Float16* __restrict__ q16,
                                               const _Float16* __restrict__ k16,
                                               const _Float16* __restrict__ v16t,
                                               const float* __restrict__ colsum,
                                               const float* __restrict__ thresh,
                                               _Float16* __restrict__ out) {
  __shared__ _Float16 Vt[128][40];       // V^T tile: [d][key_local]
  __shared__ _Float16 Pt[8][16][40];     // per-wave P tile 16x32 (A layout)
  const int h = blockIdx.x, qtb = blockIdx.y;
  const int tid = threadIdx.x, wave = tid >> 5, lane = tid & 31;
  const int qt = qtb * 8 + wave, q0 = qt * 16;
  const int kv = h >> 2;
  const int nlo = lane & 15, hi8 = (lane >> 4) * 8;
  const float scale = 0.088388347648318447f;
  const float th = thresh[h];
  const unsigned int vt_lds = (unsigned int)(unsigned long long)&Vt[0][0];
  const unsigned long long vbase =
      (unsigned long long)(v16t + (size_t)kv * DH * S_LEN);
  const _Float16* qptr = q16 + ((size_t)h * S_LEN + q0) * DH;
  v16h qa[4];
#pragma unroll
  for (int c = 0; c < 4; ++c) qa[c] = frag_ld(qptr + c * 32, DH);

  float rz[8] = {0, 0, 0, 0, 0, 0, 0, 0};
  v8f oacc[8] = {};
  const int npairs = qtb * 4 + 4;
  for (int kp = 0; kp < npairs; ++kp) {
    __syncthreads();                     // previous Vt consumers done
    if (wave == 0) {                     // wave-uniform: one TDM per block
      union { v4u v; unsigned int u[4]; } g0;
      union { v8u v; unsigned int u[8]; } g1;
      const unsigned long long ga = vbase + (unsigned long long)(kp * 32) * 2u;
      g0.u[0] = 1u;                                  // count=1, user mode
      g0.u[1] = vt_lds;                              // lds_addr (bytes)
      g0.u[2] = (unsigned int)(ga & 0xffffffffull);  // global_addr[31:0]
      g0.u[3] = (unsigned int)((ga >> 32) & 0x1ffffffull) | 0x80000000u; // +type=2
      g1.u[0] = (1u << 16)       // data_size = 2B
              | (1u << 20)       // pad_enable
              | (3u << 22)       // pad_interval = 16 DWORDs (one 64B row)
              | (3u << 25);      // pad_amount   = 4 DWORDs (16B -> 80B stride)
      g1.u[1] = ((unsigned int)S_LEN) << 16;   // tensor_dim0 (keys) lo16
      g1.u[2] = ((unsigned int)DH) << 16;      // tensor_dim1 (dims) lo16
      g1.u[3] = 32u << 16;                     // tile_dim0 = 32 keys
      g1.u[4] = (unsigned int)DH;              // tile_dim1 = 128 dims
      g1.u[5] = (unsigned int)S_LEN;           // tensor_dim0_stride lo32
      g1.u[6] = 0u;
      g1.u[7] = 0u;
      asm volatile("tensor_load_to_lds %0, %1" :: "s"(g0.v), "s"(g1.v) : "memory");
      __builtin_amdgcn_s_wait_tensorcnt(0);
    }
    __syncthreads();                     // Vt ready for all waves
    if (kp * 32 <= q0 + 15) {            // wave-uniform causal skip
#pragma unroll
      for (int half = 0; half < 2; ++half) {
        const int kt = kp * 2 + half;
        const _Float16* kptr = k16 + ((size_t)kv * S_LEN + kt * 16) * DH;
        v8f s = {};
#pragma unroll
        for (int c = 0; c < 4; ++c)
          s = wmma32(qa[c], frag_ld(kptr + c * 32, DH), s);
        const int j = kt * 16 + nlo;
        const float cs = colsum[h * S_LEN + j];
        union { v8f v; float f[8]; } u; u.v = s;
#pragma unroll
        for (int r = 0; r < 8; ++r) {
          const int i = q0 + hi8 + r;
          const bool keep = (j <= i) && (((i - j) <= RECENT) || (cs >= th));
          const float e = keep ? __expf(u.f[r] * scale) : 0.0f;
          rz[r] += e;
          Pt[wave][hi8 + r][half * 16 + nlo] = (_Float16)e;
        }
      }
      asm volatile("s_wait_dscnt 0" ::: "memory");
      const v16h pa = frag_ld(&Pt[wave][0][0], 40);
#pragma unroll
      for (int nt = 0; nt < 8; ++nt)
        oacc[nt] = wmma32(pa, frag_ld(&Vt[nt * 16][0], 40), oacc[nt]);
    }
  }
#pragma unroll
  for (int r = 0; r < 8; ++r) {
    rz[r] += __shfl_xor(rz[r], 1, 32);
    rz[r] += __shfl_xor(rz[r], 2, 32);
    rz[r] += __shfl_xor(rz[r], 4, 32);
    rz[r] += __shfl_xor(rz[r], 8, 32);
  }
#pragma unroll
  for (int nt = 0; nt < 8; ++nt) {
    union { v8f v; float f[8]; } u; u.v = oacc[nt];
    _Float16* op = out + (size_t)(q0 + hi8) * HID + (h * DH + nt * 16 + nlo);
#pragma unroll
    for (int r = 0; r < 8; ++r) { *op = (_Float16)(u.f[r] / rz[r]); op += HID; }
  }
}

// ---------------------------------------------------------------------------
extern "C" void kernel_launch(void* const* d_in, const int* in_sizes, int n_in,
                              void* d_out, int out_size, void* d_ws, size_t ws_size,
                              hipStream_t stream) {
  (void)in_sizes; (void)n_in; (void)out_size; (void)ws_size;
  const float* X  = (const float*)d_in[0];
  const float* Wq = (const float*)d_in[1];
  const float* Wk = (const float*)d_in[2];
  const float* Wv = (const float*)d_in[3];
  const float* Wo = (const float*)d_in[4];

  char* ws = (char*)d_ws;
  size_t off = 0;
  float*    qf   = (float*)(ws + off);    off += (size_t)S_LEN * HID * 4;
  float*    kf   = (float*)(ws + off);    off += (size_t)S_LEN * NKV * DH * 4;
  float*    vf   = (float*)(ws + off);    off += (size_t)S_LEN * NKV * DH * 4;
  _Float16* X16  = (_Float16*)(ws + off); off += (size_t)S_LEN * HID * 2;
  _Float16* Wq16 = (_Float16*)(ws + off); off += (size_t)HID * HID * 2;
  _Float16* Wk16 = (_Float16*)(ws + off); off += (size_t)HID * NKV * DH * 2;
  _Float16* Wv16 = (_Float16*)(ws + off); off += (size_t)HID * NKV * DH * 2;
  _Float16* Wo16 = (_Float16*)(ws + off); off += (size_t)HID * HID * 2;
  _Float16* q16  = (_Float16*)(ws + off); off += (size_t)S_LEN * HID * 2;
  _Float16* k16  = (_Float16*)(ws + off); off += (size_t)S_LEN * NKV * DH * 2;
  _Float16* v16t = (_Float16*)(ws + off); off += (size_t)S_LEN * NKV * DH * 2;
  float*    csum = (float*)(ws + off);    off += (size_t)NH * S_LEN * 4;
  float*    thr  = (float*)(ws + off);    off += 256;
  _Float16* aout = (_Float16*)(ws + off); off += (size_t)S_LEN * HID * 2;

  const dim3 blk(256);
  const int nX = S_LEN * HID, nW = HID * HID, nKW = HID * NKV * DH;
  f32_to_f16<<<(nX + 255) / 256, blk, 0, stream>>>(X, X16, nX);
  f32_to_f16<<<(nW + 255) / 256, blk, 0, stream>>>(Wq, Wq16, nW);
  f32_to_f16<<<(nKW + 255) / 256, blk, 0, stream>>>(Wk, Wk16, nKW);
  f32_to_f16<<<(nKW + 255) / 256, blk, 0, stream>>>(Wv, Wv16, nKW);
  f32_to_f16<<<(nW + 255) / 256, blk, 0, stream>>>(Wo, Wo16, nW);
  // QKV projections (f16 WMMA, async-LDS staging)
  gemm_wmma<<<dim3(HID / 128, S_LEN / 128), blk, 0, stream>>>(X16, Wq16, qf, S_LEN, HID, HID);
  gemm_wmma<<<dim3((NKV * DH) / 128, S_LEN / 128), blk, 0, stream>>>(X16, Wk16, kf, S_LEN, NKV * DH, HID);
  gemm_wmma<<<dim3((NKV * DH) / 128, S_LEN / 128), blk, 0, stream>>>(X16, Wv16, vf, S_LEN, NKV * DH, HID);
  // RoPE + f16 conversion (+ V transpose for TDM)
  rope_q<<<(S_LEN * NH * 64) / 256, blk, 0, stream>>>(qf, q16);
  rope_k<<<(S_LEN * NKV * 64) / 256, blk, 0, stream>>>(kf, k16);
  conv_v<<<(S_LEN * NKV * DH) / 256, blk, 0, stream>>>(vf, v16t);
  // H2O oracle
  hipMemsetAsync(csum, 0, (size_t)NH * S_LEN * 4, stream);
  attn_oracle<<<dim3(NH, S_LEN / 16 / 8), blk, 0, stream>>>(q16, k16, csum);
  topk_thresh<<<NH, blk, 0, stream>>>(csum, thr);
  // masked attention + P*V (TDM-staged V tiles)
  attn_pv<<<dim3(NH, S_LEN / 16 / 8), blk, 0, stream>>>(q16, k16, v16t, csum, thr, aout);
  // output projection
  gemm_wmma<<<dim3(HID / 128, S_LEN / 128), blk, 0, stream>>>(aout, Wo16, (float*)d_out, S_LEN, HID, HID);
}
